// Z100_MoE_41334765257147
// MI455X (gfx1250) — matrix operand order repaired and checked by reading, not compiled, not run on hardware
//
#include <hip/hip_runtime.h>
#include <cstdint>
#include <cstddef>

#define DIMX   2048
#define INTERX 1408
#define NEXP   16
#define TOPKX  2
#define SINTER 2816
#define TTOK   4096
#define KDIM   2048

typedef __bf16 bf16_t;
typedef __attribute__((ext_vector_type(16))) __bf16 v16bf;
typedef __attribute__((ext_vector_type(8)))  float  v8f;

// Native conversions -> v_cvt_pk_bf16_f32 / v_cvt_bf16_f32
static __device__ __forceinline__ bf16_t f2bf(float f) { return (bf16_t)f; }
static __device__ __forceinline__ uint32_t pk2(float a, float b) {
    union { bf16_t h[2]; uint32_t u; } o;
    o.h[0] = (bf16_t)a; o.h[1] = (bf16_t)b;
    return o.u;
}

// Generic pointer to a __shared__ object: low 32 bits are the LDS byte offset.
static __device__ __forceinline__ uint32_t lds_addr32(const void* p) {
    return (uint32_t)(uintptr_t)p;
}
// CDNA5 async copy: LDS[vdst] = MEM[vaddr] (16 bytes), tracked by ASYNCcnt.
static __device__ __forceinline__ void async_ld_b128(uint32_t lds_off, const void* gptr) {
    asm volatile("global_load_async_to_lds_b128 %0, %1, off"
                 :: "v"(lds_off), "v"(gptr) : "memory");
}
static __device__ __forceinline__ void wait_async0() {
    asm volatile("s_wait_asynccnt 0x0" ::: "memory");
}

// Load A fragment: lane<16 holds row r, K 0..15; lane>=16 row r, K 16..31
static __device__ __forceinline__ v16bf ld_afrag(const bf16_t* rowp, int akoff) {
    union { uint4 q[2]; v16bf v; } u;
    u.q[0] = *(const uint4*)(rowp + 2 * akoff);
    u.q[1] = *(const uint4*)(rowp + 8 + 2 * akoff);
    return u.v;
}
// Load B fragment (LDS holds weight rows): lane<16 col n, K 0..15; lane>=16 col n, K 16..31
static __device__ __forceinline__ v16bf ld_bfrag(const bf16_t* rowp, int bhalf) {
    union { uint4 q[2]; v16bf v; } u;
    const uint4* p = (const uint4*)(rowp + bhalf);
    u.q[0] = p[0]; u.q[1] = p[1];
    return u.v;
}

// ---------------- small kernels ----------------
__global__ void init_kernel(int* counts) {
    if (threadIdx.x < NEXP) counts[threadIdx.x] = 0;
}

// f32 -> bf16 bulk conversion (x and, when workspace allows, all weight tensors)
__global__ __launch_bounds__(256) void cvt_bf16_kernel(const float* __restrict__ src,
                                                       bf16_t* __restrict__ dst) {
    size_t i = ((size_t)blockIdx.x * 256 + threadIdx.x) * 8;
    float4 f0 = *(const float4*)(src + i);
    float4 f1 = *(const float4*)(src + i + 4);
    *(uint4*)(dst + i) = make_uint4(pk2(f0.x, f0.y), pk2(f0.z, f0.w),
                                    pk2(f1.x, f1.y), pk2(f1.z, f1.w));
}

__global__ __launch_bounds__(256) void gate_kernel(const float* __restrict__ x,
                                                   const float* __restrict__ gw,
                                                   int* __restrict__ counts,
                                                   int* __restrict__ ent_e,
                                                   int* __restrict__ ent_i,
                                                   float* __restrict__ ent_w) {
    int wave = threadIdx.x >> 5, lane = threadIdx.x & 31;
    int t = blockIdx.x * 8 + wave;
    const float* xr = x + (size_t)t * DIMX;
    float acc[NEXP];
#pragma unroll
    for (int e = 0; e < NEXP; ++e) acc[e] = 0.f;
    for (int k = lane; k < DIMX; k += 32) {
        float xv = xr[k];
#pragma unroll
        for (int e = 0; e < NEXP; ++e) acc[e] = fmaf(xv, gw[e * DIMX + k], acc[e]);
    }
#pragma unroll
    for (int e = 0; e < NEXP; ++e) {
        float v = acc[e];
        for (int off = 16; off > 0; off >>= 1) v += __shfl_xor(v, off, 32);
        acc[e] = v;
    }
    if (lane == 0) {
        float m = acc[0];
#pragma unroll
        for (int e = 1; e < NEXP; ++e) m = fmaxf(m, acc[e]);
        float p[NEXP]; float s = 0.f;
#pragma unroll
        for (int e = 0; e < NEXP; ++e) { p[e] = __expf(acc[e] - m); s += p[e]; }
        float inv = 1.f / s;
        int i0 = 0; float b0 = p[0];
#pragma unroll
        for (int e = 1; e < NEXP; ++e) if (p[e] > b0) { b0 = p[e]; i0 = e; }
        int i1 = -1; float b1 = -1.f;
#pragma unroll
        for (int e = 0; e < NEXP; ++e) if (e != i0 && p[e] > b1) { b1 = p[e]; i1 = e; }
        int a0 = atomicAdd(&counts[i0], 1);
        int a1 = atomicAdd(&counts[i1], 1);
        ent_e[2 * t + 0] = i0; ent_i[2 * t + 0] = a0; ent_w[2 * t + 0] = b0 * inv;
        ent_e[2 * t + 1] = i1; ent_i[2 * t + 1] = a1; ent_w[2 * t + 1] = b1 * inv;
    }
}

__global__ void offsets_kernel(const int* __restrict__ counts, int* __restrict__ offsets) {
    if (threadIdx.x == 0) {
        int s = 0;
        for (int e = 0; e < NEXP; ++e) { offsets[e] = s; s += counts[e]; }
        offsets[NEXP] = s;
    }
}

__global__ __launch_bounds__(256) void scatter_kernel(const int* __restrict__ ent_e,
                                                      const int* __restrict__ ent_i,
                                                      const float* __restrict__ ent_w,
                                                      const int* __restrict__ offsets,
                                                      int* __restrict__ s2t,
                                                      float* __restrict__ slot_w) {
    int i = blockIdx.x * 256 + threadIdx.x;
    int slot = offsets[ent_e[i]] + ent_i[i];
    s2t[slot] = i >> 1;
    slot_w[slot] = ent_w[i];
}

// ---------------- up-projection GEMM: H = silu(X@W1^T) * (X@W3^T) ----------------
template <bool GATHER, bool WBF16>
__global__ __launch_bounds__(256) void up_kernel(const bf16_t* __restrict__ Xb,
                                                 const void* __restrict__ W1,
                                                 const void* __restrict__ W3,
                                                 bf16_t* __restrict__ Hout,
                                                 const int* __restrict__ counts,
                                                 const int* __restrict__ offsets,
                                                 const int* __restrict__ s2t,
                                                 int NTOT) {
    __shared__ __align__(16) bf16_t Ab[64 * 32];
    __shared__ __align__(16) bf16_t B1[128 * 32];
    __shared__ __align__(16) bf16_t B3[128 * 32];

    int e = GATHER ? blockIdx.z : 0;
    int count = GATHER ? counts[e] : TTOK;
    int base  = GATHER ? offsets[e] : 0;
    int m0 = blockIdx.y * 64;
    if (m0 >= count) return;
    int nbase = blockIdx.x * 128;

    int tid = threadIdx.x;
    int ar = tid >> 2;            // 0..63 : A row
    int as = (tid & 3) * 8;       // 0,8,16,24 : K segment
    int arow = m0 + ar;
    const bf16_t* aSrc = nullptr;
    if (GATHER) {
        if (arow < count) aSrc = Xb + (size_t)s2t[base + arow] * KDIM + as;
    } else {
        aSrc = Xb + (size_t)arow * KDIM + as;
    }
    uint32_t ldsA = lds_addr32(&Ab[ar * 32 + as]);

    int wave = tid >> 5, lane = tid & 31;
    int wm = wave & 3, wn = wave >> 2;
    int fr = lane & 15;
    int akoff = (lane >> 4) << 3;
    int bhalf = (lane >> 4) << 4;

    v8f acc1[4] = {};
    v8f acc3[4] = {};

    for (int k0 = 0; k0 < KDIM; k0 += 32) {
        if constexpr (WBF16) {
            const bf16_t* w1b = (const bf16_t*)W1 + (size_t)e * NTOT * KDIM;
            const bf16_t* w3b = (const bf16_t*)W3 + (size_t)e * NTOT * KDIM;
            if (aSrc) async_ld_b128(ldsA, aSrc + k0);
            else      *(uint4*)&Ab[ar * 32 + as] = make_uint4(0, 0, 0, 0);
#pragma unroll
            for (int j = 0; j < 2; ++j) {
                int n = (tid >> 2) + j * 64;
                const bf16_t* p1 = w1b + (size_t)(nbase + n) * KDIM + k0 + as;
                const bf16_t* p3 = w3b + (size_t)(nbase + n) * KDIM + k0 + as;
                async_ld_b128(lds_addr32(&B1[n * 32 + as]), p1);
                async_ld_b128(lds_addr32(&B3[n * 32 + as]), p3);
                __builtin_prefetch(p1 + 32, 0, 3);
                __builtin_prefetch(p3 + 32, 0, 3);
            }
            wait_async0();
        } else {
            const float* w1e = (const float*)W1 + (size_t)e * NTOT * KDIM;
            const float* w3e = (const float*)W3 + (size_t)e * NTOT * KDIM;
            uint4 av = make_uint4(0, 0, 0, 0);
            if (aSrc) av = *(const uint4*)(aSrc + k0);
            *(uint4*)&Ab[ar * 32 + as] = av;
#pragma unroll
            for (int j = 0; j < 2; ++j) {
                int n = (tid >> 2) + j * 64;
                const float* p1 = w1e + (size_t)(nbase + n) * KDIM + k0 + as;
                const float* p3 = w3e + (size_t)(nbase + n) * KDIM + k0 + as;
                float4 f0 = *(const float4*)p1;
                float4 f1 = *(const float4*)(p1 + 4);
                *(uint4*)&B1[n * 32 + as] = make_uint4(pk2(f0.x, f0.y), pk2(f0.z, f0.w),
                                                       pk2(f1.x, f1.y), pk2(f1.z, f1.w));
                f0 = *(const float4*)p3;
                f1 = *(const float4*)(p3 + 4);
                *(uint4*)&B3[n * 32 + as] = make_uint4(pk2(f0.x, f0.y), pk2(f0.z, f0.w),
                                                       pk2(f1.x, f1.y), pk2(f1.z, f1.w));
                __builtin_prefetch(p1 + 32, 0, 3);
                __builtin_prefetch(p3 + 32, 0, 3);
            }
        }
        __syncthreads();

        v16bf af = ld_afrag(&Ab[(wm * 16 + fr) * 32], akoff);
        v16bf bf1[4], bf3[4];
#pragma unroll
        for (int nt = 0; nt < 4; ++nt) {
            bf1[nt] = ld_bfrag(&B1[(wn * 64 + nt * 16 + fr) * 32], bhalf);
            bf3[nt] = ld_bfrag(&B3[(wn * 64 + nt * 16 + fr) * 32], bhalf);
        }
#pragma unroll
        for (int nt = 0; nt < 4; ++nt) {
            acc1[nt] = __builtin_amdgcn_wmma_f32_16x16x32_bf16(false, af, false, bf1[nt],
                                                               (short)0, acc1[nt], false, false);
            acc3[nt] = __builtin_amdgcn_wmma_f32_16x16x32_bf16(false, af, false, bf3[nt],
                                                               (short)0, acc3[nt], false, false);
        }
        __syncthreads();
    }

    int col = lane & 15;
    int rb = (lane >> 4) << 3;
#pragma unroll
    for (int nt = 0; nt < 4; ++nt) {
#pragma unroll
        for (int v = 0; v < 8; ++v) {
            int ml = m0 + wm * 16 + rb + v;
            if (ml < count) {
                float u = acc1[nt][v];
                float g = u / (1.f + __expf(-u));
                float h = g * acc3[nt][v];
                Hout[(size_t)(base + ml) * NTOT + nbase + wn * 64 + nt * 16 + col] = f2bf(h);
            }
        }
    }
}

// ---------------- down-projection GEMM: Out (+)= (H @ W2^T) [* slot weight] ----------------
template <int MODE, bool WBF16>  // MODE 0 = direct store (shared), 1 = scaled atomic scatter (routed)
__global__ __launch_bounds__(256) void down_kernel(const bf16_t* __restrict__ Hin,
                                                   const void* __restrict__ W2,
                                                   float* __restrict__ Out,
                                                   const int* __restrict__ counts,
                                                   const int* __restrict__ offsets,
                                                   const int* __restrict__ s2t,
                                                   const float* __restrict__ slot_w,
                                                   int K) {
    __shared__ __align__(16) bf16_t Ab[64 * 32];
    __shared__ __align__(16) bf16_t Bb[128 * 32];

    int e = (MODE == 1) ? blockIdx.z : 0;
    int count = (MODE == 1) ? counts[e] : TTOK;
    int base  = (MODE == 1) ? offsets[e] : 0;
    int m0 = blockIdx.y * 64;
    if (m0 >= count) return;
    int nbase = blockIdx.x * 128;

    int tid = threadIdx.x;
    int ar = tid >> 2;
    int as = (tid & 3) * 8;
    int arow = m0 + ar;
    const bf16_t* aSrc = (arow < count) ? (Hin + (size_t)(base + arow) * K + as) : nullptr;
    uint32_t ldsA = lds_addr32(&Ab[ar * 32 + as]);

    int wave = tid >> 5, lane = tid & 31;
    int wm = wave & 3, wn = wave >> 2;
    int fr = lane & 15;
    int akoff = (lane >> 4) << 3;
    int bhalf = (lane >> 4) << 4;

    v8f acc[4] = {};

    for (int k0 = 0; k0 < K; k0 += 32) {
        if constexpr (WBF16) {
            const bf16_t* w2b = (const bf16_t*)W2 + (size_t)e * DIMX * K;
            if (aSrc) async_ld_b128(ldsA, aSrc + k0);
            else      *(uint4*)&Ab[ar * 32 + as] = make_uint4(0, 0, 0, 0);
#pragma unroll
            for (int j = 0; j < 2; ++j) {
                int n = (tid >> 2) + j * 64;
                const bf16_t* p = w2b + (size_t)(nbase + n) * K + k0 + as;
                async_ld_b128(lds_addr32(&Bb[n * 32 + as]), p);
                __builtin_prefetch(p + 32, 0, 3);
            }
            wait_async0();
        } else {
            const float* w2e = (const float*)W2 + (size_t)e * DIMX * K;
            uint4 av = make_uint4(0, 0, 0, 0);
            if (aSrc) av = *(const uint4*)(aSrc + k0);
            *(uint4*)&Ab[ar * 32 + as] = av;
#pragma unroll
            for (int j = 0; j < 2; ++j) {
                int n = (tid >> 2) + j * 64;
                const float* p = w2e + (size_t)(nbase + n) * K + k0 + as;
                float4 f0 = *(const float4*)p;
                float4 f1 = *(const float4*)(p + 4);
                *(uint4*)&Bb[n * 32 + as] = make_uint4(pk2(f0.x, f0.y), pk2(f0.z, f0.w),
                                                       pk2(f1.x, f1.y), pk2(f1.z, f1.w));
                __builtin_prefetch(p + 32, 0, 3);
            }
        }
        __syncthreads();

        v16bf af = ld_afrag(&Ab[(wm * 16 + fr) * 32], akoff);
        v16bf bfr[4];
#pragma unroll
        for (int nt = 0; nt < 4; ++nt)
            bfr[nt] = ld_bfrag(&Bb[(wn * 64 + nt * 16 + fr) * 32], bhalf);
#pragma unroll
        for (int nt = 0; nt < 4; ++nt)
            acc[nt] = __builtin_amdgcn_wmma_f32_16x16x32_bf16(false, af, false, bfr[nt],
                                                              (short)0, acc[nt], false, false);
        __syncthreads();
    }

    int col = lane & 15;
    int rb = (lane >> 4) << 3;
#pragma unroll
    for (int nt = 0; nt < 4; ++nt) {
#pragma unroll
        for (int v = 0; v < 8; ++v) {
            int ml = m0 + wm * 16 + rb + v;
            int n = nbase + wn * 64 + nt * 16 + col;
            if (MODE == 0) {
                Out[(size_t)ml * DIMX + n] = acc[nt][v];
            } else if (ml < count) {
                int slot = base + ml;
                unsafeAtomicAdd(&Out[(size_t)s2t[slot] * DIMX + n], slot_w[slot] * acc[nt][v]);
            }
        }
    }
}

// ---------------- host side ----------------
extern "C" void kernel_launch(void* const* d_in, const int* in_sizes, int n_in,
                              void* d_out, int out_size, void* d_ws, size_t ws_size,
                              hipStream_t stream) {
    (void)in_sizes; (void)n_in; (void)out_size;
    const float* x   = (const float*)d_in[0];
    const float* gw  = (const float*)d_in[1];
    const float* w1  = (const float*)d_in[2];
    const float* w2  = (const float*)d_in[3];
    const float* w3  = (const float*)d_in[4];
    const float* sw1 = (const float*)d_in[5];
    const float* sw2 = (const float*)d_in[6];
    const float* sw3 = (const float*)d_in[7];
    float* out = (float*)d_out;

    char* ws = (char*)d_ws;
    int*    counts  = (int*)  (ws + 0);
    int*    offsets = (int*)  (ws + 256);
    int*    ent_e   = (int*)  (ws + 512);
    int*    ent_i   = (int*)  (ws + 33280);
    float*  ent_w   = (float*)(ws + 66048);
    int*    s2t     = (int*)  (ws + 98816);
    float*  slot_w  = (float*)(ws + 131584);
    bf16_t* xb      = (bf16_t*)(ws + 164352);
    bf16_t* H       = (bf16_t*)(ws + 164352 + (size_t)TTOK * DIMX * 2);
    bf16_t* Hs      = (bf16_t*)(ws + 164352 + (size_t)TTOK * DIMX * 2
                                 + (size_t)TTOK * TOPKX * INTERX * 2);

    // optional bf16 weight cache region
    const size_t NW1 = (size_t)NEXP * INTERX * KDIM;   // w1 / w3 elems
    const size_t NW2 = (size_t)NEXP * DIMX * INTERX;   // w2 elems
    const size_t NS1 = (size_t)SINTER * KDIM;          // sw1 / sw3 elems
    const size_t NS2 = (size_t)DIMX * SINTER;          // sw2 elems
    size_t o = 63078912ull;
    bf16_t* w1b  = (bf16_t*)(ws + o); o += NW1 * 2;
    bf16_t* w2b  = (bf16_t*)(ws + o); o += NW2 * 2;
    bf16_t* w3b  = (bf16_t*)(ws + o); o += NW1 * 2;
    bf16_t* sw1b = (bf16_t*)(ws + o); o += NS1 * 2;
    bf16_t* sw2b = (bf16_t*)(ws + o); o += NS2 * 2;
    bf16_t* sw3b = (bf16_t*)(ws + o); o += NS1 * 2;
    const bool big = ws_size >= o;

    init_kernel<<<1, 32, 0, stream>>>(counts);
    cvt_bf16_kernel<<<(TTOK * DIMX / 8) / 256, 256, 0, stream>>>(x, xb);
    gate_kernel<<<TTOK / 8, 256, 0, stream>>>(x, gw, counts, ent_e, ent_i, ent_w);
    offsets_kernel<<<1, 32, 0, stream>>>(counts, offsets);
    scatter_kernel<<<(TTOK * TOPKX) / 256, 256, 0, stream>>>(ent_e, ent_i, ent_w,
                                                             offsets, s2t, slot_w);
    if (big) {
        cvt_bf16_kernel<<<NW1 / 8 / 256, 256, 0, stream>>>(w1,  w1b);
        cvt_bf16_kernel<<<NW2 / 8 / 256, 256, 0, stream>>>(w2,  w2b);
        cvt_bf16_kernel<<<NW1 / 8 / 256, 256, 0, stream>>>(w3,  w3b);
        cvt_bf16_kernel<<<NS1 / 8 / 256, 256, 0, stream>>>(sw1, sw1b);
        cvt_bf16_kernel<<<NS2 / 8 / 256, 256, 0, stream>>>(sw2, sw2b);
        cvt_bf16_kernel<<<NS1 / 8 / 256, 256, 0, stream>>>(sw3, sw3b);
        // shared experts
        up_kernel<false, true><<<dim3(SINTER / 128, TTOK / 64, 1), 256, 0, stream>>>(
            xb, sw1b, sw3b, Hs, counts, offsets, s2t, SINTER);
        down_kernel<0, true><<<dim3(DIMX / 128, TTOK / 64, 1), 256, 0, stream>>>(
            Hs, sw2b, out, counts, offsets, s2t, slot_w, SINTER);
        // routed experts
        up_kernel<true, true><<<dim3(INTERX / 128, TTOK / 64, NEXP), 256, 0, stream>>>(
            xb, w1b, w3b, H, counts, offsets, s2t, INTERX);
        down_kernel<1, true><<<dim3(DIMX / 128, TTOK / 64, NEXP), 256, 0, stream>>>(
            H, w2b, out, counts, offsets, s2t, slot_w, INTERX);
    } else {
        up_kernel<false, false><<<dim3(SINTER / 128, TTOK / 64, 1), 256, 0, stream>>>(
            xb, sw1, sw3, Hs, counts, offsets, s2t, SINTER);
        down_kernel<0, false><<<dim3(DIMX / 128, TTOK / 64, 1), 256, 0, stream>>>(
            Hs, sw2, out, counts, offsets, s2t, slot_w, SINTER);
        up_kernel<true, false><<<dim3(INTERX / 128, TTOK / 64, NEXP), 256, 0, stream>>>(
            xb, w1, w3, H, counts, offsets, s2t, INTERX);
        down_kernel<1, false><<<dim3(DIMX / 128, TTOK / 64, NEXP), 256, 0, stream>>>(
            H, w2, out, counts, offsets, s2t, slot_w, INTERX);
    }
}
